// SaShiMiMamba_27427661153051
// MI455X (gfx1250) — matrix operand order repaired and checked by reading, not compiled, not run on hardware
//
#include <hip/hip_runtime.h>
#include <math.h>

// ---------------------------------------------------------------------------
// Types for CDNA5 WMMA (wave32)
// ---------------------------------------------------------------------------
typedef __attribute__((ext_vector_type(16))) __bf16 v16bf;
typedef __attribute__((ext_vector_type(8)))  __bf16 v8bf;
typedef __attribute__((ext_vector_type(8)))  float  v8f;

__device__ __forceinline__ __bf16 f2bf(float f) {
    unsigned u = __builtin_bit_cast(unsigned, f);
    unsigned short s = (unsigned short)((u + 0x7FFFu + ((u >> 16) & 1u)) >> 16);
    return __builtin_bit_cast(__bf16, s);
}
__device__ __forceinline__ float siluf(float x) { return x / (1.f + expf(-x)); }
__device__ __forceinline__ float softplusf(float x) {
    return (x > 20.f) ? x : log1pf(expf(x));
}

#define ACT_NONE 0
#define ACT_SOFTPLUS 1

#define BM 64
#define BN 64
#define BK 32

// ---------------------------------------------------------------------------
// GEMM: C[M,N] = act(A[M,K(lda)] @ W[N,K]^T + bias[N])
// Contract (satisfied by every call in this network):
//   - M is a multiple of 64 (grid.x divides exactly; no m-store guard)
//   - K is a multiple of 8 (tail chunks are all-in or all-out)
// 256 threads = 8 waves; 64x64 tile; each wave computes 16x32 (two WMMA
// accumulators sharing one A fragment). Out-of-range rows are handled by
// CLAMPING the row pointer (not predication): garbage A-rows / B-rows only
// feed never-stored outputs. Zero-fill happens only in the K direction (tail).
// Main loop is branch-free: 2x2 global_load_b128 -> bf16 -> ds_store_b128,
// fragment gathers are contiguous ds_load_b128 per the ISA 16-bit
// A(16x32)/B(32x16) lane layouts. EXEC is all-ones at every WMMA.
// ---------------------------------------------------------------------------
__global__ __launch_bounds__(256)
void gemm_wmma_bf16(const float* __restrict__ A, int lda,
                    const float* __restrict__ W,
                    const float* __restrict__ bias,
                    float* __restrict__ C, int ldc,
                    int M, int N, int K, int act)
{
    __shared__ __bf16 As[BM][BK];
    __shared__ __bf16 Bs[BN][BK];

    const int tid  = threadIdx.x;
    const int lane = tid & 31;
    const int wave = tid >> 5;        // 0..7
    const int wm   = wave & 3;        // row sub-tile (16 rows each)
    const int wn   = wave >> 2;       // col group (32 cols each)
    const int half = lane >> 4;
    const int l15  = lane & 15;

    const int m0 = blockIdx.x * BM;
    const int n0 = blockIdx.y * BN;

    // loader assignment: thread -> (row 0..63, 8-wide col chunk)
    const int lrow = tid >> 2;
    const int lcol = (tid & 3) * 8;

    // clamped row pointers: no load predication needed for M/N edges
    const int arow = m0 + lrow;
    const int brow = n0 + lrow;
    const float* Arow = A + (size_t)((arow < M) ? arow : (M - 1)) * lda;
    const float* Wrow = W + (size_t)((brow < N) ? brow : (N - 1)) * K;

    v8f acc0 = {}, acc1 = {};

    const int kFull = K & ~(BK - 1);
    int k0 = 0;

    // ---------------- branch-free main loop over full K tiles ----------------
    for (; k0 < kFull; k0 += BK) {
        const int kb = k0 + lcol;
        {
            float4 p0 = *(const float4*)(Arow + kb);
            float4 p1 = *(const float4*)(Arow + kb + 4);
            v8bf v;
            v[0] = f2bf(p0.x); v[1] = f2bf(p0.y); v[2] = f2bf(p0.z); v[3] = f2bf(p0.w);
            v[4] = f2bf(p1.x); v[5] = f2bf(p1.y); v[6] = f2bf(p1.z); v[7] = f2bf(p1.w);
            *(v8bf*)&As[lrow][lcol] = v;
        }
        {
            float4 p0 = *(const float4*)(Wrow + kb);
            float4 p1 = *(const float4*)(Wrow + kb + 4);
            v8bf v;
            v[0] = f2bf(p0.x); v[1] = f2bf(p0.y); v[2] = f2bf(p0.z); v[3] = f2bf(p0.w);
            v[4] = f2bf(p1.x); v[5] = f2bf(p1.y); v[6] = f2bf(p1.z); v[7] = f2bf(p1.w);
            *(v8bf*)&Bs[lrow][lcol] = v;
        }
        // speculative prefetch of next K tile (gfx1250 global_prefetch_b8);
        // invalid addresses are silently dropped per ISA, so no guard needed.
        __builtin_prefetch(Arow + kb + BK, 0, 3);
        __builtin_prefetch(Wrow + kb + BK, 0, 3);
        __syncthreads();

        // A: lane row = wm*16+l15; elems 0..7 -> k = half*8+e ; 8..15 -> 16+half*8+e
        v8bf alo = *(const v8bf*)&As[wm * 16 + l15][half * 8];
        v8bf ahi = *(const v8bf*)&As[wm * 16 + l15][16 + half * 8];
        v16bf af = __builtin_shufflevector(alo, ahi,
                      0, 1, 2, 3, 4, 5, 6, 7, 8, 9, 10, 11, 12, 13, 14, 15);
        // B: elems 0..7 -> k = half*8+e ; 8..15 -> 16+half*8+e (same as A)
        v8bf b0lo = *(const v8bf*)&Bs[wn * 32 + l15][half * 8];
        v8bf b0hi = *(const v8bf*)&Bs[wn * 32 + l15][16 + half * 8];
        v16bf bf0 = __builtin_shufflevector(b0lo, b0hi,
                      0, 1, 2, 3, 4, 5, 6, 7, 8, 9, 10, 11, 12, 13, 14, 15);
        v8bf b1lo = *(const v8bf*)&Bs[wn * 32 + 16 + l15][half * 8];
        v8bf b1hi = *(const v8bf*)&Bs[wn * 32 + 16 + l15][16 + half * 8];
        v16bf bf1 = __builtin_shufflevector(b1lo, b1hi,
                      0, 1, 2, 3, 4, 5, 6, 7, 8, 9, 10, 11, 12, 13, 14, 15);

        acc0 = __builtin_amdgcn_wmma_f32_16x16x32_bf16(false, af, false, bf0,
                                                       (short)0, acc0, false, false);
        acc1 = __builtin_amdgcn_wmma_f32_16x16x32_bf16(false, af, false, bf1,
                                                       (short)0, acc1, false, false);
        __syncthreads();
    }

    // ---------------- one tail tile for K % 32 != 0 (K multiple of 8) --------
    if (k0 < K) {
        const int kb = k0 + lcol;
        {
            float f[8] = {0.f, 0.f, 0.f, 0.f, 0.f, 0.f, 0.f, 0.f};
            if (kb + 8 <= K) {
                float4 p0 = *(const float4*)(Arow + kb);
                float4 p1 = *(const float4*)(Arow + kb + 4);
                f[0] = p0.x; f[1] = p0.y; f[2] = p0.z; f[3] = p0.w;
                f[4] = p1.x; f[5] = p1.y; f[6] = p1.z; f[7] = p1.w;
            }
            v8bf v;
#pragma unroll
            for (int e = 0; e < 8; ++e) v[e] = f2bf(f[e]);
            *(v8bf*)&As[lrow][lcol] = v;
        }
        {
            float f[8] = {0.f, 0.f, 0.f, 0.f, 0.f, 0.f, 0.f, 0.f};
            if (kb + 8 <= K) {
                float4 p0 = *(const float4*)(Wrow + kb);
                float4 p1 = *(const float4*)(Wrow + kb + 4);
                f[0] = p0.x; f[1] = p0.y; f[2] = p0.z; f[3] = p0.w;
                f[4] = p1.x; f[5] = p1.y; f[6] = p1.z; f[7] = p1.w;
            }
            v8bf v;
#pragma unroll
            for (int e = 0; e < 8; ++e) v[e] = f2bf(f[e]);
            *(v8bf*)&Bs[lrow][lcol] = v;
        }
        __syncthreads();

        v8bf alo = *(const v8bf*)&As[wm * 16 + l15][half * 8];
        v8bf ahi = *(const v8bf*)&As[wm * 16 + l15][16 + half * 8];
        v16bf af = __builtin_shufflevector(alo, ahi,
                      0, 1, 2, 3, 4, 5, 6, 7, 8, 9, 10, 11, 12, 13, 14, 15);
        v8bf b0lo = *(const v8bf*)&Bs[wn * 32 + l15][half * 8];
        v8bf b0hi = *(const v8bf*)&Bs[wn * 32 + l15][16 + half * 8];
        v16bf bf0 = __builtin_shufflevector(b0lo, b0hi,
                      0, 1, 2, 3, 4, 5, 6, 7, 8, 9, 10, 11, 12, 13, 14, 15);
        v8bf b1lo = *(const v8bf*)&Bs[wn * 32 + 16 + l15][half * 8];
        v8bf b1hi = *(const v8bf*)&Bs[wn * 32 + 16 + l15][16 + half * 8];
        v16bf bf1 = __builtin_shufflevector(b1lo, b1hi,
                      0, 1, 2, 3, 4, 5, 6, 7, 8, 9, 10, 11, 12, 13, 14, 15);

        acc0 = __builtin_amdgcn_wmma_f32_16x16x32_bf16(false, af, false, bf0,
                                                       (short)0, acc0, false, false);
        acc1 = __builtin_amdgcn_wmma_f32_16x16x32_bf16(false, af, false, bf1,
                                                       (short)0, acc1, false, false);
    }

    // ---- epilogue (M multiple of 64 => no m guard; n guard for ragged N) ----
#pragma unroll
    for (int t = 0; t < 2; ++t) {
        const v8f acc = t ? acc1 : acc0;
        const int n = n0 + wn * 32 + t * 16 + l15;
        if (n >= N) continue;
        const float bv = bias ? bias[n] : 0.f;
        float* Cp = C + (size_t)(m0 + wm * 16 + (half << 3)) * ldc + n;
#pragma unroll
        for (int i = 0; i < 8; ++i) {
            float v = acc[i] + bv;
            if (act == ACT_SOFTPLUS) v = softplusf(v);
            Cp[(size_t)i * ldc] = v;
        }
    }
}

// ---------------------------------------------------------------------------
// Elementwise / helper kernels
// ---------------------------------------------------------------------------
__global__ void embed_kernel(const int* __restrict__ tok,
                             const float* __restrict__ emb,
                             float* __restrict__ out, int rows, int dm)
{
    int idx = blockIdx.x * blockDim.x + threadIdx.x;
    if (idx >= rows * dm) return;
    int r = idx / dm, c = idx % dm;
    out[idx] = emb[(size_t)tok[r] * dm + c];
}

__global__ void add_inplace_kernel(float* __restrict__ y,
                                   const float* __restrict__ x, int n)
{
    int idx = blockIdx.x * blockDim.x + threadIdx.x;
    if (idx < n) y[idx] += x[idx];
}

__global__ void rmsnorm_kernel(const float* __restrict__ x,
                               const float* __restrict__ w,
                               float* __restrict__ out, int m)
{
    __shared__ float red[256];
    const int row = blockIdx.x;
    const float* xr = x + (size_t)row * m;
    float s = 0.f;
    for (int c = threadIdx.x; c < m; c += blockDim.x) { float v = xr[c]; s += v * v; }
    red[threadIdx.x] = s;
    __syncthreads();
    for (int off = 128; off; off >>= 1) {
        if (threadIdx.x < off) red[threadIdx.x] += red[threadIdx.x + off];
        __syncthreads();
    }
    const float scale = rsqrtf(red[0] / (float)m + 1e-5f);
    for (int c = threadIdx.x; c < m; c += blockDim.x)
        out[(size_t)row * m + c] = xr[c] * scale * w[c];
}

// causal depthwise conv (4 taps) + SiLU; x is the first `din` columns of xz
__global__ void conv_silu_kernel(const float* __restrict__ xz,
                                 const float* __restrict__ cw,
                                 const float* __restrict__ cb,
                                 float* __restrict__ xc,
                                 int Bn, int Lc, int din)
{
    int idx = blockIdx.x * blockDim.x + threadIdx.x;
    if (idx >= Bn * Lc * din) return;
    int d = idx % din;
    int l = (idx / din) % Lc;
    int b = idx / (din * Lc);
    const int stride = 2 * din;
    const float* base = xz + (size_t)b * Lc * stride + d;
    float acc = cb[d];
#pragma unroll
    for (int k = 0; k < 4; ++k) {
        int ls = l + k - 3;
        if (ls >= 0) acc += base[(size_t)ls * stride] * cw[d * 4 + k];
    }
    xc[idx] = siluf(acc);
}

// selective scan: one lane per (batch, channel); 16 fp32 states in registers
__global__ void scan_kernel(const float* __restrict__ xdbl, int nx, int dtr,
                            const float* __restrict__ dt,
                            const float* __restrict__ xc,
                            const float* __restrict__ xz,
                            const float* __restrict__ A_log,
                            const float* __restrict__ Dp,
                            float* __restrict__ y,
                            int Bn, int Lc, int din)
{
    int idx = blockIdx.x * blockDim.x + threadIdx.x;
    if (idx >= Bn * din) return;
    const int d = idx % din, b = idx / din;
    float Ar[16], h[16];
#pragma unroll
    for (int s = 0; s < 16; ++s) { Ar[s] = -expf(A_log[(size_t)d * 16 + s]); h[s] = 0.f; }
    const float Dd = Dp[d];
    for (int l = 0; l < Lc; ++l) {
        const size_t r = (size_t)b * Lc + l;
        const float dtv = dt[r * din + d];
        const float xcv = xc[r * din + d];
        const float* xr = xdbl + r * nx + dtr;
        float accv = 0.f;
#pragma unroll
        for (int s = 0; s < 16; ++s) {
            float da = expf(dtv * Ar[s]);
            h[s] = da * h[s] + dtv * xr[s] * xcv;
            accv += h[s] * xr[16 + s];
        }
        const float zv = xz[r * (size_t)(2 * din) + din + d];
        y[r * din + d] = (accv + Dd * xcv) * siluf(zv);
    }
}

// upsample: shift u one step along Ln (zero-fill first), reshape to (LnOut,Hout), add residual
__global__ void shift_reshape_add(const float* __restrict__ u,
                                  const float* __restrict__ res,
                                  float* __restrict__ out,
                                  int Bn, int Ln, int PHoQ, int Hout)
{
    const int LnOut = Ln * PHoQ / Hout;
    int idx = blockIdx.x * blockDim.x + threadIdx.x;
    if (idx >= Bn * LnOut * Hout) return;
    int c  = idx % Hout;
    int lo = (idx / Hout) % LnOut;
    int b  = idx / (Hout * LnOut);
    int pos  = lo * Hout + c;
    int urow = pos / PHoQ, ucol = pos % PHoQ;
    float v = (urow == 0) ? 0.f : u[((size_t)b * Ln + urow - 1) * PHoQ + ucol];
    out[idx] = v + res[idx];
}

// ---------------------------------------------------------------------------
// Host orchestration
// ---------------------------------------------------------------------------
namespace {

struct StageP {
    const float *A_log, *D, *conv_b, *conv_w, *dt_b, *dt_w, *in_w, *norm_f, *norm_w, *out_w, *x_w;
    int m;
};

StageP mkstage(void* const* din, int base, int m) {
    StageP s;
    s.A_log  = (const float*)din[base + 0];
    s.D      = (const float*)din[base + 1];
    s.conv_b = (const float*)din[base + 2];
    s.conv_w = (const float*)din[base + 3];
    s.dt_b   = (const float*)din[base + 4];
    s.dt_w   = (const float*)din[base + 5];
    s.in_w   = (const float*)din[base + 6];
    s.norm_f = (const float*)din[base + 7];
    s.norm_w = (const float*)din[base + 8];
    s.out_w  = (const float*)din[base + 9];
    s.x_w    = (const float*)din[base + 10];
    s.m = m;
    return s;
}

inline void gemm(const float* A, int lda, const float* W, const float* bias,
                 float* C, int ldc, int M, int N, int K, int act, hipStream_t st)
{
    dim3 grid((M + BM - 1) / BM, (N + BN - 1) / BN);
    gemm_wmma_bf16<<<grid, dim3(256), 0, st>>>(A, lda, W, bias, C, ldc, M, N, K, act);
}

inline void eadd(float* y, const float* x, int n, hipStream_t st) {
    add_inplace_kernel<<<(n + 255) / 256, 256, 0, st>>>(y, x, n);
}

struct Bufs { float *res, *hid, *nrm, *xz, *xc, *xdbl, *dt, *y; };

// Run one Mamba stage over T rows of width m: xout = stage(xin). xout may alias xin.
void run_stage(const StageP& sp, const float* xin, float* xout, int T, int Bn,
               const Bufs& w, hipStream_t st)
{
    const int m = sp.m, din = 2 * m, dtr = m / 16, nx = dtr + 32;
    const int Lc = T / Bn;
    const int NL = 2;
    for (int l = 0; l < NL; ++l) {
        if (l == 0)
            hipMemcpyAsync(w.res, xin, (size_t)T * m * sizeof(float),
                           hipMemcpyDeviceToDevice, st);
        else
            eadd(w.res, w.hid, T * m, st);
        rmsnorm_kernel<<<T, 256, 0, st>>>(w.res, sp.norm_w + (size_t)l * m, w.nrm, m);
        // in_proj: (T,m) x (4m,m)^T -> (T,4m)  [x | z]
        gemm(w.nrm, m, sp.in_w + (size_t)l * 4 * m * m, nullptr,
             w.xz, 2 * din, T, 2 * din, m, ACT_NONE, st);
        // causal conv + silu
        {
            int n = T * din;
            conv_silu_kernel<<<(n + 255) / 256, 256, 0, st>>>(
                w.xz, sp.conv_w + (size_t)l * din * 4, sp.conv_b + (size_t)l * din,
                w.xc, Bn, Lc, din);
        }
        // x_proj: (T,din) x (nx,din)^T -> (T,nx)
        gemm(w.xc, din, sp.x_w + (size_t)l * nx * din, nullptr,
             w.xdbl, nx, T, nx, din, ACT_NONE, st);
        // dt_proj + softplus: (T,dtr slice of xdbl) x (din,dtr)^T -> (T,din)
        gemm(w.xdbl, nx, sp.dt_w + (size_t)l * din * dtr, sp.dt_b + (size_t)l * din,
             w.dt, din, T, din, dtr, ACT_SOFTPLUS, st);
        // selective scan -> y
        {
            int n = Bn * din;
            scan_kernel<<<(n + 255) / 256, 256, 0, st>>>(
                w.xdbl, nx, dtr, w.dt, w.xc, w.xz,
                sp.A_log + (size_t)l * din * 16, sp.D + (size_t)l * din,
                w.y, Bn, Lc, din);
        }
        // out_proj: (T,din) x (m,din)^T -> hid (T,m)
        gemm(w.y, din, sp.out_w + (size_t)l * m * din, nullptr,
             w.hid, m, T, m, din, ACT_NONE, st);
    }
    eadd(w.hid, w.res, T * m, st);
    rmsnorm_kernel<<<T, 256, 0, st>>>(w.hid, sp.norm_f, xout, m);
}

} // namespace

extern "C" void kernel_launch(void* const* d_in, const int* in_sizes, int n_in,
                              void* d_out, int out_size, void* d_ws, size_t ws_size,
                              hipStream_t stream)
{
    (void)in_sizes; (void)n_in; (void)out_size; (void)ws_size;
    const int Bn = 2, Lseq = 2048, DM = 128, VOC = 256;

    // d_in in JAX pytree order (sorted dict keys, lists by index):
    // down[0].{b,w}, down[1].{b,w}, embedding,
    // stages_down[0..2] (11 leaves each), stages_up[0..1] (11 leaves each),
    // up[0].{b,w}, up[1].{b,w}, tokens
    const float* down0_b = (const float*)d_in[0];
    const float* down0_w = (const float*)d_in[1];
    const float* down1_b = (const float*)d_in[2];
    const float* down1_w = (const float*)d_in[3];
    const float* emb     = (const float*)d_in[4];
    StageP sd0 = mkstage(d_in, 5,  128);
    StageP sd1 = mkstage(d_in, 16, 256);
    StageP sd2 = mkstage(d_in, 27, 512);
    StageP su0 = mkstage(d_in, 38, 128);
    StageP su1 = mkstage(d_in, 49, 256);
    const float* up0_b = (const float*)d_in[60];
    const float* up0_w = (const float*)d_in[61];
    const float* up1_b = (const float*)d_in[62];
    const float* up1_w = (const float*)d_in[63];
    const int*   tok   = (const int*)d_in[64];

    // Workspace bump allocator (floats)
    float* ws = (float*)d_ws;
    size_t off = 0;
    auto alloc = [&](size_t n) { float* p = ws + off; off += (n + 63) & ~(size_t)63; return p; };
    Bufs wb;
    wb.res  = alloc(524288);
    wb.hid  = alloc(524288);
    wb.nrm  = alloc(524288);
    wb.xz   = alloc(2097152);
    wb.xc   = alloc(1048576);
    wb.xdbl = alloc(262144);
    wb.dt   = alloc(1048576);
    wb.y    = alloc(1048576);
    float* xA = alloc(524288);
    float* xB = alloc(524288);
    float* ub = alloc(524288);
    float* r0 = alloc(524288);
    float* r1 = alloc(262144);
    float* r2 = alloc(131072);

    const int T0 = Bn * Lseq;  // 4096 rows @ 128

    // Embedding gather
    embed_kernel<<<(T0 * DM + 255) / 256, 256, 0, stream>>>(tok, emb, xA, T0, DM);

    // ---- down path ----
    hipMemcpyAsync(r0, xA, (size_t)T0 * DM * sizeof(float), hipMemcpyDeviceToDevice, stream);
    run_stage(sd0, xA, xA, T0, Bn, wb, stream);                       // (4096,128)
    gemm(xA, 512, down0_w, down0_b, xB, 256, 1024, 256, 512, ACT_NONE, stream); // (1024,256)

    hipMemcpyAsync(r1, xB, (size_t)1024 * 256 * sizeof(float), hipMemcpyDeviceToDevice, stream);
    run_stage(sd1, xB, xB, 1024, Bn, wb, stream);                     // (1024,256)
    gemm(xB, 1024, down1_w, down1_b, xA, 512, 256, 512, 1024, ACT_NONE, stream); // (256,512)

    hipMemcpyAsync(r2, xA, (size_t)256 * 512 * sizeof(float), hipMemcpyDeviceToDevice, stream);
    run_stage(sd2, xA, xA, 256, Bn, wb, stream);                      // (256,512)
    eadd(xA, r2, 256 * 512, stream);

    // ---- up path s=1 ----
    gemm(xA, 512, up1_w, up1_b, ub, 1024, 256, 1024, 512, ACT_NONE, stream);    // (256,1024)
    {
        int n = Bn * 512 * 256;
        shift_reshape_add<<<(n + 255) / 256, 256, 0, stream>>>(ub, r1, xB, Bn, 128, 1024, 256);
    }
    run_stage(su1, xB, xB, 1024, Bn, wb, stream);                     // (1024,256)

    // ---- up path s=0 ----
    gemm(xB, 256, up0_w, up0_b, ub, 512, 1024, 512, 256, ACT_NONE, stream);     // (1024,512)
    {
        int n = Bn * 2048 * 128;
        shift_reshape_add<<<(n + 255) / 256, 256, 0, stream>>>(ub, r0, xA, Bn, 512, 512, 128);
    }
    run_stage(su0, xA, xA, T0, Bn, wb, stream);                       // (4096,128)

    // ---- logits: x @ embedding^T -> d_out (4096,256) ----
    gemm(xA, DM, emb, nullptr, (float*)d_out, VOC, T0, VOC, DM, ACT_NONE, stream);
}
